// BlockMemory_trainableFalse_1297080123749
// MI455X (gfx1250) — matrix operand order repaired and checked by reading, not compiled, not run on hardware
//
#include <hip/hip_runtime.h>

// Problem constants (from reference): query (32,256,64,64) f32, mem (512,256) f32
#define N_PIX      131072
#define M_SLOTS    512
#define D_FEAT     256
#define TILE_R     64
#define QHAT_ELEMS 33554432   // 32*256*64*64

typedef __attribute__((ext_vector_type(16))) __bf16       v16bf;
typedef __attribute__((ext_vector_type(8)))  float        v8f;
typedef __attribute__((ext_vector_type(4)))  unsigned int v4u;

union Frag { v16bf bf; v4u u[2]; };
union Acc  { v8f v; float f[8]; };

__device__ inline unsigned short f2bf(float x) {
  unsigned int u = __float_as_uint(x);
  return (unsigned short)((u + 0x7fffu + ((u >> 16) & 1u)) >> 16);  // RNE
}

__device__ inline v8f wmma_bf16(v16bf a, v16bf b, v8f c) {
  return __builtin_amdgcn_wmma_f32_16x16x32_bf16(false, a, false, b, (short)0, c, false, false);
}

// A fragment (16x32 bf16) from LDS row-major [rows][lda].
// lanes 0-15: M=lane, K chunks [kBase+0..7],[kBase+16..23];
// lanes 16-31: M=lane-16, K chunks [kBase+8..15],[kBase+24..31].
__device__ inline v16bf load_a(const unsigned short* base, int lda, int rowBase, int kBase, int lane) {
  int half = lane >> 4;
  const unsigned short* p = base + (rowBase + (lane & 15)) * lda + kBase + half * 8;
  Frag f;
  f.u[0] = *(const v4u*)p;
  f.u[1] = *(const v4u*)(p + 16);
  return f.bf;
}

// B fragment (32x16 bf16): element (k, n) = src[(colBase+n)*lda + kBase + k].
// lanes 0-15: column n=lane, K chunks [kBase+0..7],[kBase+16..23]; lanes 16-31: n=lane-16, K chunks [kBase+8..15],[kBase+24..31].
__device__ inline v16bf load_b(const unsigned short* src, int lda, int kBase, int colBase, int lane) {
  int half = lane >> 4;
  const unsigned short* p = src + (colBase + (lane & 15)) * lda + kBase + half * 8;
  Frag f;
  f.u[0] = *(const v4u*)p;
  f.u[1] = *(const v4u*)(p + 16);
  return f.bf;
}

__device__ inline void atomicMaxF(float* addr, float val) {
  unsigned int* ua = (unsigned int*)addr;
  unsigned int old = *ua;
  while (true) {
    if (__uint_as_float(old) >= val) break;
    unsigned int assumed = old;
    old = atomicCAS(ua, assumed, __float_as_uint(val));
    if (old == assumed) break;
  }
}

// ---------- K0: convert mem to bf16 (row-major + transposed), init scratch ----------
__global__ void k_init(const float* __restrict__ mem, unsigned short* __restrict__ mem_bf,
                       unsigned short* __restrict__ memT_bf, float* __restrict__ agg,
                       float* __restrict__ slotRowMax, int* __restrict__ count) {
  int s = blockIdx.x, c = threadIdx.x;
  unsigned short h = f2bf(mem[s * D_FEAT + c]);
  mem_bf[s * D_FEAT + c] = h;
  memT_bf[c * M_SLOTS + s] = h;
  agg[s * D_FEAT + c] = 0.0f;
  if (c == 0) { slotRowMax[s] = -3.402823466e38f; count[s] = 0; }
}

// ---------- K1: fused score GEMM + row softmax + threshold/renorm + read GEMM ----------
// 512 threads = 16 waves; wave -> 16 rows x 128 cols in GEMM1 (8 accumulators = 64 VGPRs)
// to stay spill-free; 4 column-quarter waves combine softmax stats through LDS.
__global__ void __launch_bounds__(512) k_main(
    const float* __restrict__ q, const unsigned short* __restrict__ mem_bf,
    const unsigned short* __restrict__ memT_bf, float* __restrict__ out_qhat,
    float* __restrict__ rowmax_g, int* __restrict__ nearest_g,
    float* __restrict__ slotRowMax, int* __restrict__ count) {
  extern __shared__ char smem[];
  unsigned short* Abuf = (unsigned short*)smem;              // [64][264] bf16 q tile
  unsigned short* Pbuf = (unsigned short*)(smem + 33792);    // [64][520] bf16 probs
  float* sMax  = (float*)(smem + 100352);                    // [4][64]
  int*   sIdx  = (int*)  (smem + 101376);                    // [4][64]
  float* sSum  = (float*)(smem + 102400);                    // [4][64]
  float* sKept = (float*)(smem + 103424);                    // [4][64]
  float* Qout  = (float*)smem;                               // [64][261] alias (post-barrier)

  const int t = threadIdx.x;
  const int lane = t & 31;
  const int wave = t >> 5;
  const int half = lane >> 4;
  const int n0 = blockIdx.x * TILE_R;
  const int bimg = n0 >> 12;
  const int rem = n0 & 4095;
  const long long gbase = (long long)bimg * (D_FEAT * 4096) + rem;

  // Load q tile (strided over channel dim; lanes contiguous over pixels) -> bf16 LDS
  {
    int pix = t & 63, cg = t >> 6;               // cg in 0..7
    for (int cc = 0; cc < 32; ++cc) {
      int c = cg * 32 + cc;
      Abuf[pix * 264 + c] = f2bf(q[gbase + (long long)c * 4096 + pix]);
    }
  }
  __syncthreads();

  // GEMM1: score[64x512] = q_tile(bf16) @ mem(bf16)^T, fp32 accumulate.
  // wave -> 16 rows x 128 cols (column quarter), K = 256.
  const int rowBase  = (wave & 3) * 16;
  const int ch       = wave >> 2;                // column quarter 0..3
  const int colBase0 = ch * 128;
  Acc acc[8];
  #pragma unroll
  for (int i = 0; i < 8; ++i)
    #pragma unroll
    for (int j = 0; j < 8; ++j) acc[i].f[j] = 0.0f;

  for (int kk = 0; kk < 8; ++kk) {
    v16bf a = load_a(Abuf, 264, rowBase, kk * 32, lane);
    #pragma unroll
    for (int ct = 0; ct < 8; ++ct) {
      v16bf b = load_b(mem_bf, D_FEAT, kk * 32, colBase0 + ct * 16, lane);
      acc[ct].v = wmma_bf16(a, b, acc[ct].v);
    }
  }

  // Row stats. C layout: VGPR j holds (row = rowBase + half*8 + j, col = tile*16 + lane&15).
  float rmax[8], rsum[8], kept[8], gM[8], gS[8];
  int ridx[8];
  #pragma unroll
  for (int j = 0; j < 8; ++j) { rmax[j] = -3.402823466e38f; ridx[j] = 0; }
  for (int ct = 0; ct < 8; ++ct) {
    int col = colBase0 + ct * 16 + (lane & 15);
    #pragma unroll
    for (int j = 0; j < 8; ++j) {
      float v = acc[ct].f[j];
      if (v > rmax[j]) { rmax[j] = v; ridx[j] = col; }
    }
  }
  #pragma unroll
  for (int m = 1; m < 16; m <<= 1)
    #pragma unroll
    for (int j = 0; j < 8; ++j) {
      float om = __shfl_xor(rmax[j], m, 32);
      int   oi = __shfl_xor(ridx[j], m, 32);
      if (om > rmax[j] || (om == rmax[j] && oi < ridx[j])) { rmax[j] = om; ridx[j] = oi; }
    }
  #pragma unroll
  for (int j = 0; j < 8; ++j) rsum[j] = 0.0f;
  for (int ct = 0; ct < 8; ++ct)
    #pragma unroll
    for (int j = 0; j < 8; ++j) rsum[j] += __expf(acc[ct].f[j] - rmax[j]);
  #pragma unroll
  for (int m = 1; m < 16; m <<= 1)
    #pragma unroll
    for (int j = 0; j < 8; ++j) rsum[j] += __shfl_xor(rsum[j], m, 32);

  if ((lane & 15) == 0)
    #pragma unroll
    for (int j = 0; j < 8; ++j) {
      int r = rowBase + half * 8 + j;
      sMax[ch * 64 + r] = rmax[j];
      sIdx[ch * 64 + r] = ridx[j];
      sSum[ch * 64 + r] = rsum[j];
    }
  __syncthreads();

  #pragma unroll
  for (int j = 0; j < 8; ++j) {
    int r = rowBase + half * 8 + j;
    float M = sMax[r];
    for (int qq = 1; qq < 4; ++qq) M = fmaxf(M, sMax[qq * 64 + r]);
    float S = 0.0f;
    for (int qq = 0; qq < 4; ++qq) S += sSum[qq * 64 + r] * __expf(sMax[qq * 64 + r] - M);
    gM[j] = M;
    gS[j] = S;
  }

  // Threshold (p < 1/m -> 0) and L1 renorm sum of kept weights.
  const float thr = 1.0f / (float)M_SLOTS;
  #pragma unroll
  for (int j = 0; j < 8; ++j) kept[j] = 0.0f;
  for (int ct = 0; ct < 8; ++ct)
    #pragma unroll
    for (int j = 0; j < 8; ++j) {
      float p = __expf(acc[ct].f[j] - gM[j]) / gS[j];
      if (p >= thr) kept[j] += p;
    }
  #pragma unroll
  for (int m = 1; m < 16; m <<= 1)
    #pragma unroll
    for (int j = 0; j < 8; ++j) kept[j] += __shfl_xor(kept[j], m, 32);
  if ((lane & 15) == 0)
    #pragma unroll
    for (int j = 0; j < 8; ++j) sKept[ch * 64 + rowBase + half * 8 + j] = kept[j];
  __syncthreads();
  #pragma unroll
  for (int j = 0; j < 8; ++j) {
    int r = rowBase + half * 8 + j;
    kept[j] = fmaxf(sKept[r] + sKept[64 + r] + sKept[128 + r] + sKept[192 + r], 1e-12f);
  }

  // Renormalized thresholded probabilities -> LDS bf16 (A operand of GEMM2)
  for (int ct = 0; ct < 8; ++ct) {
    int col = colBase0 + ct * 16 + (lane & 15);
    #pragma unroll
    for (int j = 0; j < 8; ++j) {
      int r = rowBase + half * 8 + j;
      float p = __expf(acc[ct].f[j] - gM[j]) / gS[j];
      p = (p >= thr) ? (p / kept[j]) : 0.0f;
      Pbuf[r * 520 + col] = f2bf(p);
    }
  }

  // Per-pixel side outputs for the update path (colsum/colmax cancel analytically).
  if (ch == 0 && (lane & 15) == 0)
    for (int j = 0; j < 8; ++j) {
      int r = rowBase + half * 8 + j;
      float M = sMax[r];
      int ni = sIdx[r];
      for (int qq = 1; qq < 4; ++qq) {
        float mq = sMax[qq * 64 + r];
        if (mq > M) { M = mq; ni = sIdx[qq * 64 + r]; }
      }
      rowmax_g[n0 + r] = gM[j];
      nearest_g[n0 + r] = ni;
      atomicMaxF(&slotRowMax[ni], gM[j]);
      atomicAdd(&count[ni], 1);
    }
  __syncthreads();

  // GEMM2: query_hat[64x256] = probs(bf16) @ mem(bf16), K = 512.
  // wave -> 16 rows x 64 cols (column quarter of d), 4 accumulators.
  const int rowBase2 = (wave & 3) * 16;
  const int cBase0   = (wave >> 2) * 64;
  Acc acc2[4];
  #pragma unroll
  for (int i = 0; i < 4; ++i)
    #pragma unroll
    for (int j = 0; j < 8; ++j) acc2[i].f[j] = 0.0f;
  for (int kk = 0; kk < 16; ++kk) {
    v16bf a = load_a(Pbuf, 520, rowBase2, kk * 32, lane);
    #pragma unroll
    for (int ct = 0; ct < 4; ++ct) {
      v16bf b = load_b(memT_bf, M_SLOTS, kk * 32, cBase0 + ct * 16, lane);
      acc2[ct].v = wmma_bf16(a, b, acc2[ct].v);
    }
  }
  __syncthreads();  // all Pbuf/Abuf reads done before aliasing with Qout

  for (int ct = 0; ct < 4; ++ct) {
    int c = cBase0 + ct * 16 + (lane & 15);
    #pragma unroll
    for (int j = 0; j < 8; ++j)
      Qout[(rowBase2 + half * 8 + j) * 261 + c] = acc2[ct].f[j];
  }
  __syncthreads();

  // Store transposed back to [b,d,h,w]: lanes contiguous over pixels (w) -> coalesced.
  {
    int pix = t & 63, cg = t >> 6;
    for (int cc = 0; cc < 32; ++cc) {
      int c = cg * 32 + cc;
      out_qhat[gbase + (long long)c * 4096 + pix] = Qout[pix * 261 + c];
    }
  }
}

// ---------- K2: sparse cluster scatter-add: agg[s] += exp(rowmax - slotRowMax[s]) * q[n] ----------
__global__ void __launch_bounds__(256) k_agg(
    const float* __restrict__ q, const float* __restrict__ rowmax_g,
    const int* __restrict__ nearest_g, const float* __restrict__ slotRowMax,
    float* __restrict__ agg) {
  __shared__ float wgt[64];
  __shared__ int   slt[64];
  int t = threadIdx.x;
  int n0 = blockIdx.x * 64;
  if (t < 64) {
    int n = n0 + t;
    int s = nearest_g[n];
    slt[t] = s;
    wgt[t] = __expf(rowmax_g[n] - slotRowMax[s]);
  }
  __syncthreads();
  int pix = t & 63, cg = t >> 6;
  int bimg = n0 >> 12, rem = n0 & 4095;
  long long gbase = (long long)bimg * (D_FEAT * 4096) + rem;
  int s = slt[pix];
  float w = wgt[pix];
  for (int cc = 0; cc < 64; ++cc) {
    int c = cg * 64 + cc;
    float v = q[gbase + (long long)c * 4096 + pix];
    atomicAdd(&agg[s * D_FEAT + c], w * v);
  }
}

// ---------- K3: mem_update = L2-normalize(mem + agg), zero empty slots ----------
__global__ void __launch_bounds__(256) k_final(
    const float* __restrict__ mem, const float* __restrict__ agg,
    const int* __restrict__ count, float* __restrict__ out_mem) {
  __shared__ float red[256];
  int s = blockIdx.x, c = threadIdx.x;
  float u = mem[s * D_FEAT + c] + agg[s * D_FEAT + c];
  red[c] = u * u;
  __syncthreads();
  for (int off = 128; off > 0; off >>= 1) {
    if (c < off) red[c] += red[c + off];
    __syncthreads();
  }
  float nrm = sqrtf(red[0]);
  out_mem[s * D_FEAT + c] = (count[s] > 0) ? (u / fmaxf(nrm, 1e-12f)) : 0.0f;
}

extern "C" void kernel_launch(void* const* d_in, const int* in_sizes, int n_in,
                              void* d_out, int out_size, void* d_ws, size_t ws_size,
                              hipStream_t stream) {
  (void)in_sizes; (void)n_in; (void)out_size; (void)ws_size;
  const float* q   = (const float*)d_in[0];
  const float* mem = (const float*)d_in[1];
  float* out_qhat  = (float*)d_out;
  float* out_mem   = out_qhat + QHAT_ELEMS;

  // Workspace layout (~2.1 MB total)
  char* ws = (char*)d_ws;
  float*          rowmax     = (float*)ws;                       // N f32
  int*            nearest    = (int*)(ws + 524288);              // N i32
  float*          slotRowMax = (float*)(ws + 1048576);           // 512 f32
  int*            count      = (int*)(ws + 1050624);             // 512 i32
  float*          agg        = (float*)(ws + 1052672);           // 512*256 f32
  unsigned short* mem_bf     = (unsigned short*)(ws + 1576960);  // 512*256 bf16
  unsigned short* memT_bf    = (unsigned short*)(ws + 1839104);  // 256*512 bf16

  k_init <<<M_SLOTS, D_FEAT, 0, stream>>>(mem, mem_bf, memT_bf, agg, slotRowMax, count);
  k_main <<<N_PIX / TILE_R, 512, 104448, stream>>>(q, mem_bf, memT_bf, out_qhat,
                                                   rowmax, nearest, slotRowMax, count);
  k_agg  <<<N_PIX / 64, 256, 0, stream>>>(q, rowmax, nearest, slotRowMax, agg);
  k_final<<<M_SLOTS, D_FEAT, 0, stream>>>(mem, agg, count, out_mem);
}
